// HeteroGNN_45079976739433
// MI455X (gfx1250) — matrix operand order repaired and checked by reading, not compiled, not run on hardware
//
#include <hip/hip_runtime.h>
#include <hip/hip_bf16.h>

// ---------------------------------------------------------------------------
// HeteroGNN (2-layer GAT) for MI455X / gfx1250, wave32, WMMA f16->f32.
// Memory-bound problem: WMMA GEMMs for projections (f16 operands staged via
// LDS), atomic segment-softmax for the edge aggregation.
// ---------------------------------------------------------------------------

#define NC_   100000
#define NI_   1000
#define F_IN_ 128
#define EMB_  32
#define HID_  128
#define H_    2
#define HC_   256          // H_ * HID_
#define BN_EPS_ 1e-5f

typedef __attribute__((ext_vector_type(16))) _Float16 v16h;
typedef __attribute__((ext_vector_type(8)))  _Float16 v8h;
typedef __attribute__((ext_vector_type(8)))  float    v8f;

#define LDSB_STRIDE 40   // halves per LDS row (80B: 16B-aligned, bank-staggered)

// ---------------------------------------------------------------------------
// f32 -> f16 conversion (pre-pass so the GEMM streams pure f16)
// ---------------------------------------------------------------------------
__global__ void cvt_f32_to_f16(const float* __restrict__ src,
                               _Float16* __restrict__ dst, int n) {
    int i = blockIdx.x * blockDim.x + threadIdx.x;
    if (i < n) dst[i] = (_Float16)src[i];
}

// ---------------------------------------------------------------------------
// WMMA GEMM: C[M x N] = A[M x K] * B[K x N] (+ bias[N]); A,B f16 row-major,
// C f32. One wave owns a 16(M) x 64(N) tile; 4 waves/block => 64 rows.
// B k-tile (32x64) is staged transposed in LDS so each lane reads its
// 16-half WMMA B fragment with two ds_load_b128.  All four B fragments are
// loaded into distinct registers before the four WMMAs so the ds_loads
// overlap and the WMMAs issue back-to-back.
// Lane layouts per CDNA5 ISA 7.12.2 (wave32):
//   A 16x32 f16: lane ll=lane&15 -> M=ll; half e: k=(e>=8?16:0)+(lane>>4)*8+(e&7)
//   B 32x16 f16: lane ll -> N=ll; half e: k=(e>=8?16:0)+(lane>>4)*8+(e&7)
//   C/D 16x16 f32: lane ll -> N=ll; vgpr r -> M=r+8*(lane>>4)
// ---------------------------------------------------------------------------
__global__ __launch_bounds__(128) void gemm_wmma_f16(
    const _Float16* __restrict__ A,   // [M x K]
    const _Float16* __restrict__ B,   // [K x N]
    float* __restrict__ C, int ldc,
    int M, int N, int K,
    const float* __restrict__ bias)
{
    __shared__ __align__(16) _Float16 ldsB[64 * LDSB_STRIDE];

    const int lane = threadIdx.x & 31;
    const int wave = threadIdx.x >> 5;
    const int ll   = lane & 15;
    const int hi   = lane >> 4;
    const int tid  = threadIdx.x;

    const int m0 = (blockIdx.x * 4 + wave) * 16;
    const int n0 = blockIdx.y * 64;

    v8f acc[4];
#pragma unroll
    for (int t = 0; t < 4; ++t)
#pragma unroll
        for (int i = 0; i < 8; ++i) acc[t][i] = 0.0f;

    const int  arow    = m0 + ll;
    const bool arow_ok = (arow < M);
    const _Float16* Arow = A + (size_t)(arow_ok ? arow : 0) * K;

    // staging assignment: thread -> (k row, 16-col segment) of the 32x64 tile
    const int skk = tid >> 2;          // 0..31
    const int sn  = (tid & 3) * 16;    // 0,16,32,48

    for (int k0 = 0; k0 < K; k0 += 32) {
        __syncthreads();   // protect previous iteration's LDS reads
        {
            const _Float16* brow = B + (size_t)(k0 + skk) * N + (n0 + sn);
            const v8h b0 = *(const v8h*)(brow);
            const v8h b1 = *(const v8h*)(brow + 8);
#pragma unroll
            for (int j = 0; j < 8; ++j) {
                ldsB[(sn + j)     * LDSB_STRIDE + skk] = b0[j];
                ldsB[(sn + 8 + j) * LDSB_STRIDE + skk] = b1[j];
            }
        }
        __syncthreads();

        // ---- A fragment: two contiguous b128 loads of 8 halves
        v16h a;
        if (arow_ok) {
            const v8h a0 = *(const v8h*)(Arow + k0 + hi * 8);
            const v8h a1 = *(const v8h*)(Arow + k0 + 16 + hi * 8);
            a = __builtin_shufflevector(a0, a1,
                0, 1, 2, 3, 4, 5, 6, 7, 8, 9, 10, 11, 12, 13, 14, 15);
        } else {
#pragma unroll
            for (int j = 0; j < 16; ++j) a[j] = (_Float16)0.0f;
        }

        // ---- load all 4 B fragments first (8 ds_load_b128 in flight) ...
        v16h bfrag[4];
#pragma unroll
        for (int t = 0; t < 4; ++t) {
            const int bn = t * 16 + ll;                    // column within tile
            const v8h* bp0 = (const v8h*)&ldsB[bn * LDSB_STRIDE + hi * 8];
            const v8h* bp1 = (const v8h*)&ldsB[bn * LDSB_STRIDE + 16 + hi * 8];
            bfrag[t] = __builtin_shufflevector(bp0[0], bp1[0],
                0, 1, 2, 3, 4, 5, 6, 7, 8, 9, 10, 11, 12, 13, 14, 15);
        }
        // ---- ... then issue the 4 WMMAs back-to-back (no mutual hazards)
#pragma unroll
        for (int t = 0; t < 4; ++t) {
            acc[t] = __builtin_amdgcn_wmma_f32_16x16x32_f16(
                false, a, false, bfrag[t], (short)0, acc[t], false, false);
        }
    }

#pragma unroll
    for (int t = 0; t < 4; ++t) {
        const int bn = n0 + t * 16 + ll;
        const float bv = bias ? bias[bn] : 0.0f;
#pragma unroll
        for (int r = 0; r < 8; ++r) {
            const int m = m0 + r + 8 * hi;
            if (m < M) C[(size_t)m * ldc + bn] = acc[t][r] + bv;
        }
    }
}

// ---------------------------------------------------------------------------
// Small elementwise / reduction kernels
// ---------------------------------------------------------------------------
__global__ void fill_f32(float* __restrict__ p, float v, int n) {
    int i = blockIdx.x * blockDim.x + threadIdx.x;
    if (i < n) p[i] = v;
}
__global__ void fill_u32(unsigned* __restrict__ p, unsigned v, int n) {
    int i = blockIdx.x * blockDim.x + threadIdx.x;
    if (i < n) p[i] = v;
}

// xi = emb_table[x_industry] @ W_ep (32x128) + b_ep
__global__ void xi_proj_kernel(const float* __restrict__ emb,
                               const int* __restrict__ xind,
                               const float* __restrict__ W,
                               const float* __restrict__ b,
                               float* __restrict__ xi) {
    int t = blockIdx.x * blockDim.x + threadIdx.x;
    if (t >= NI_ * 128) return;
    const int n = t >> 7, c = t & 127;
    const int row = xind[n];
    float acc = b[c];
    const float* er = emb + (size_t)row * EMB_;
#pragma unroll
    for (int k = 0; k < EMB_; ++k) acc += er[k] * W[k * 128 + c];
    xi[t] = acc;
}

// v_dst[h*128+k] = sum_c W_dst[k, h*128+c] * att_dst[h*128+c]   (256 threads)
__global__ void vdst_kernel(const float* __restrict__ Wd,
                            const float* __restrict__ att,
                            float* __restrict__ v) {
    int t = blockIdx.x * blockDim.x + threadIdx.x;
    if (t >= 256) return;
    const int h = t >> 7, k = t & 127;
    float acc = 0.f;
    const float* wr = Wd + (size_t)k * HC_ + h * 128;
    const float* ar = att + h * 128;
    for (int c = 0; c < 128; ++c) acc += wr[c] * ar[c];
    v[h * 128 + k] = acc;
}

// a_d[n,h] = x_dst[n,:] . v[h,:]
__global__ void ad_kernel(const float* __restrict__ x,
                          const float* __restrict__ v,
                          float* __restrict__ a_d, int N) {
    int n = blockIdx.x * blockDim.x + threadIdx.x;
    if (n >= N) return;
    float a0 = 0.f, a1 = 0.f;
    const float* xr = x + (size_t)n * 128;
    for (int k = 0; k < 128; ++k) {
        const float xv = xr[k];
        a0 += xv * v[k];
        a1 += xv * v[128 + k];
    }
    a_d[n * 2 + 0] = a0;
    a_d[n * 2 + 1] = a1;
}

// a_s[n,h] = hs[n,h,:] . att_src[h,:]
__global__ void as_kernel(const float* __restrict__ hs,
                          const float* __restrict__ att,
                          float* __restrict__ a_s, int N) {
    int t = blockIdx.x * blockDim.x + threadIdx.x;
    if (t >= N * 2) return;
    const int n = t >> 1, h = t & 1;
    float acc = 0.f;
    const float* hr = hs + (size_t)n * HC_ + h * 128;
    const float* ar = att + h * 128;
    for (int c = 0; c < 128; ++c) acc += hr[c] * ar[c];
    a_s[t] = acc;
}

// ---------------------------------------------------------------------------
// Segment softmax over dst (ordered-uint float atomic max, then exp/sum/agg)
// ---------------------------------------------------------------------------
__device__ __forceinline__ unsigned f2ord(float f) {
    unsigned u = __float_as_uint(f);
    return (u & 0x80000000u) ? ~u : (u | 0x80000000u);
}
__device__ __forceinline__ float ord2f(unsigned s) {
    return __uint_as_float((s & 0x80000000u) ? (s ^ 0x80000000u) : ~s);
}
__device__ __forceinline__ float leaky02(float l) {
    return (l < 0.f) ? 0.2f * l : l;
}

__global__ void edge_max_kernel(const int* __restrict__ src, const int* __restrict__ dst,
                                int E, const float* __restrict__ a_s,
                                const float* __restrict__ a_d,
                                unsigned* __restrict__ mbuf) {
    int e = blockIdx.x * blockDim.x + threadIdx.x;
    if (e >= E) return;
    const int s = src[e], d = dst[e];
#pragma unroll
    for (int h = 0; h < 2; ++h) {
        const float l = leaky02(a_s[s * 2 + h] + a_d[d * 2 + h]);
        atomicMax(&mbuf[d * 2 + h], f2ord(l));
    }
}

__global__ void edge_exp_kernel(const int* __restrict__ src, const int* __restrict__ dst,
                                int E, const float* __restrict__ a_s,
                                const float* __restrict__ a_d,
                                const unsigned* __restrict__ mbuf,
                                float* __restrict__ denom,
                                float* __restrict__ ew) {
    int e = blockIdx.x * blockDim.x + threadIdx.x;
    if (e >= E) return;
    const int s = src[e], d = dst[e];
#pragma unroll
    for (int h = 0; h < 2; ++h) {
        const float l  = leaky02(a_s[s * 2 + h] + a_d[d * 2 + h]);
        const float m  = ord2f(mbuf[d * 2 + h]);
        const float ex = expf(l - m);
        ew[e * 2 + h] = ex;
        atomicAdd(&denom[d * 2 + h], ex);
    }
}

// One 256-thread block per edge: agg[dst, c] += 0.5 * w_h * hs[src, h, c]
// (0.5 = head-mean folded into the weight; heads h=0/1 map to j<128 / j>=128)
__global__ void edge_aggr_kernel(const int* __restrict__ src, const int* __restrict__ dst,
                                 int E, const float* __restrict__ ew,
                                 const float* __restrict__ denom,
                                 const float* __restrict__ hs,
                                 float* __restrict__ agg) {
    const int e = blockIdx.x;
    if (e >= E) return;
    const int j = threadIdx.x;          // 0..255
    const int h = j >> 7;
    const int s = src[e], d = dst[e];
    const float w = 0.5f * ew[e * 2 + h] / (denom[d * 2 + h] + 1e-16f);
    atomicAdd(&agg[(size_t)d * 128 + (j & 127)], w * hs[(size_t)s * HC_ + j]);
}

__global__ void add_bias_kernel(float* __restrict__ agg, const float* __restrict__ bias, int N) {
    int t = blockIdx.x * blockDim.x + threadIdx.x;
    if (t >= N * 128) return;
    agg[t] += bias[t & 127];
}

// x = relu(bn_eval(agg)) + x      (in-place residual update)
__global__ void bn_relu_res_kernel(const float* __restrict__ agg,
                                   const float* __restrict__ g, const float* __restrict__ b,
                                   const float* __restrict__ mean, const float* __restrict__ var,
                                   float* __restrict__ x, int N) {
    int t = blockIdx.x * blockDim.x + threadIdx.x;
    if (t >= N * 128) return;
    const int c = t & 127;
    float v = (agg[t] - mean[c]) * rsqrtf(var[c] + BN_EPS_) * g[c] + b[c];
    v = v > 0.f ? v : 0.f;
    x[t] += v;
}

__global__ void lin_kernel(const float* __restrict__ x, const float* __restrict__ W,
                           const float* __restrict__ b, float* __restrict__ out, int N) {
    int n = blockIdx.x * blockDim.x + threadIdx.x;
    if (n >= N) return;
    float acc = 0.f;
    const float* xr = x + (size_t)n * 128;
    for (int k = 0; k < 128; ++k) acc += xr[k] * W[k];
    out[n] = acc + b[0];
}

// ---------------------------------------------------------------------------
// Host orchestration
// ---------------------------------------------------------------------------
static void run_gemm(const float* A, const float* B, float* C,
                     int M, int N, int K, const float* bias,
                     _Float16* Ah, _Float16* Bh, hipStream_t s) {
    const int an = M * K;
    cvt_f32_to_f16<<<(an + 255) / 256, 256, 0, s>>>(A, Ah, an);
    const int bn = K * N;
    cvt_f32_to_f16<<<(bn + 255) / 256, 256, 0, s>>>(B, Bh, bn);
    dim3 gg((M + 63) / 64, N / 64);
    gemm_wmma_f16<<<gg, 128, 0, s>>>(Ah, Bh, C, N, M, N, K, bias);
}

struct Conv {
    const float* x_src; int Ns;
    const float* x_dst; int Nd;
    const int* src; const int* dst; int E;
    const float* Wsrc; const float* Wdst;
    const float* atts; const float* attd;
    const float* bias;
    float* agg;
};

static void run_conv(const Conv& c, float* hs, float* a_s, float* a_d,
                     unsigned* mbuf, float* denom, float* ew, float* vbuf,
                     _Float16* Ah, _Float16* Bh, hipStream_t s) {
    // hs[Ns, 256] = x_src @ W_src    (WMMA)
    run_gemm(c.x_src, c.Wsrc, hs, c.Ns, HC_, HID_, nullptr, Ah, Bh, s);
    as_kernel<<<(c.Ns * 2 + 255) / 256, 256, 0, s>>>(hs, c.atts, a_s, c.Ns);
    vdst_kernel<<<1, 256, 0, s>>>(c.Wdst, c.attd, vbuf);
    ad_kernel<<<(c.Nd + 255) / 256, 256, 0, s>>>(c.x_dst, vbuf, a_d, c.Nd);

    fill_u32<<<(c.Nd * 2 + 255) / 256, 256, 0, s>>>(mbuf, 0u, c.Nd * 2);
    fill_f32<<<(c.Nd * 2 + 255) / 256, 256, 0, s>>>(denom, 0.f, c.Nd * 2);

    edge_max_kernel<<<(c.E + 255) / 256, 256, 0, s>>>(c.src, c.dst, c.E, a_s, a_d, mbuf);
    edge_exp_kernel<<<(c.E + 255) / 256, 256, 0, s>>>(c.src, c.dst, c.E, a_s, a_d,
                                                      mbuf, denom, ew);
    edge_aggr_kernel<<<c.E, 256, 0, s>>>(c.src, c.dst, c.E, ew, denom, hs, c.agg);
    add_bias_kernel<<<(c.Nd * 128 + 255) / 256, 256, 0, s>>>(c.agg, c.bias, c.Nd);
}

static inline size_t align_up(size_t x) { return (x + 255) & ~(size_t)255; }

extern "C" void kernel_launch(void* const* d_in, const int* in_sizes, int n_in,
                              void* d_out, int out_size, void* d_ws, size_t ws_size,
                              hipStream_t stream) {
    // ---- inputs (setup_inputs dict insertion order, params flattened DFS) ----
    const float* x_company  = (const float*)d_in[0];
    const int*   x_industry = (const int*)  d_in[1];
    const int* ecc_s = (const int*)d_in[2]; const int* ecc_d = (const int*)d_in[3];
    const int* eci_s = (const int*)d_in[4]; const int* eci_d = (const int*)d_in[5];
    const int* eic_s = (const int*)d_in[6]; const int* eic_d = (const int*)d_in[7];
    const int E_cc = in_sizes[2], E_ci = in_sizes[4], E_ic = in_sizes[6];

    const float* fpW = (const float*)d_in[8];
    const float* fpB = (const float*)d_in[9];
    const float* epW = (const float*)d_in[10];
    const float* epB = (const float*)d_in[11];
    const float* emb = (const float*)d_in[12];
    const float* linW = (const float*)d_in[59];
    const float* linB = (const float*)d_in[60];
    (void)n_in; (void)out_size; (void)ws_size;

    // ---- workspace carve ----
    char* p = (char*)d_ws;
    auto carve = [&](size_t bytes) { void* r = (void*)p; p += align_up(bytes); return r; };
    float*    xc    = (float*)    carve((size_t)NC_ * HID_ * 4);
    float*    xi    = (float*)    carve((size_t)NI_ * HID_ * 4);
    float*    agg_c = (float*)    carve((size_t)NC_ * HID_ * 4);
    float*    agg_i = (float*)    carve((size_t)NI_ * HID_ * 4);
    float*    hs    = (float*)    carve((size_t)NC_ * HC_ * 4);
    float*    a_s   = (float*)    carve((size_t)NC_ * H_ * 4);
    float*    a_d   = (float*)    carve((size_t)NC_ * H_ * 4);
    unsigned* mbuf  = (unsigned*) carve((size_t)NC_ * H_ * 4);
    float*    denom = (float*)    carve((size_t)NC_ * H_ * 4);
    float*    ew    = (float*)    carve((size_t)1000000 * H_ * 4);
    float*    vbuf  = (float*)    carve(256 * 4);
    _Float16* Ah    = (_Float16*) carve((size_t)NC_ * HID_ * 2);
    _Float16* Bh    = (_Float16*) carve((size_t)HID_ * HC_ * 2);

    // ---- input projections ----
    run_gemm(x_company, fpW, xc, NC_, HID_, F_IN_, fpB, Ah, Bh, stream);
    xi_proj_kernel<<<(NI_ * 128 + 255) / 256, 256, 0, stream>>>(emb, x_industry, epW, epB, xi);

    // ---- 2 hetero-GAT layers ----
    for (int L = 0; L < 2; ++L) {
        const int base = 13 + L * 23;
        const float* ccWs = (const float*)d_in[base + 0];
        const float* ccWd = (const float*)d_in[base + 1];
        const float* ccAs = (const float*)d_in[base + 2];
        const float* ccAd = (const float*)d_in[base + 3];
        const float* ccB  = (const float*)d_in[base + 4];
        const float* icWs = (const float*)d_in[base + 5];
        const float* icWd = (const float*)d_in[base + 6];
        const float* icAs = (const float*)d_in[base + 7];
        const float* icAd = (const float*)d_in[base + 8];
        const float* icB  = (const float*)d_in[base + 9];
        const float* ciWs = (const float*)d_in[base + 10];
        const float* ciWd = (const float*)d_in[base + 11];
        const float* ciAs = (const float*)d_in[base + 12];
        const float* ciAd = (const float*)d_in[base + 13];
        const float* ciB  = (const float*)d_in[base + 14];
        const float* bncG = (const float*)d_in[base + 15];
        const float* bncB = (const float*)d_in[base + 16];
        const float* bncM = (const float*)d_in[base + 17];
        const float* bncV = (const float*)d_in[base + 18];
        const float* bniG = (const float*)d_in[base + 19];
        const float* bniB = (const float*)d_in[base + 20];
        const float* bniM = (const float*)d_in[base + 21];
        const float* bniV = (const float*)d_in[base + 22];

        fill_f32<<<(NC_ * 128 + 255) / 256, 256, 0, stream>>>(agg_c, 0.f, NC_ * 128);
        fill_f32<<<(NI_ * 128 + 255) / 256, 256, 0, stream>>>(agg_i, 0.f, NI_ * 128);

        Conv cc{xc, NC_, xc, NC_, ecc_s, ecc_d, E_cc, ccWs, ccWd, ccAs, ccAd, ccB, agg_c};
        run_conv(cc, hs, a_s, a_d, mbuf, denom, ew, vbuf, Ah, Bh, stream);
        Conv ic{xi, NI_, xc, NC_, eic_s, eic_d, E_ic, icWs, icWd, icAs, icAd, icB, agg_c};
        run_conv(ic, hs, a_s, a_d, mbuf, denom, ew, vbuf, Ah, Bh, stream);
        Conv ci{xc, NC_, xi, NI_, eci_s, eci_d, E_ci, ciWs, ciWd, ciAs, ciAd, ciB, agg_i};
        run_conv(ci, hs, a_s, a_d, mbuf, denom, ew, vbuf, Ah, Bh, stream);

        bn_relu_res_kernel<<<(NC_ * 128 + 255) / 256, 256, 0, stream>>>(
            agg_c, bncG, bncB, bncM, bncV, xc, NC_);
        bn_relu_res_kernel<<<(NI_ * 128 + 255) / 256, 256, 0, stream>>>(
            agg_i, bniG, bniB, bniM, bniV, xi, NI_);
    }

    // ---- output head ----
    lin_kernel<<<(NC_ + 255) / 256, 256, 0, stream>>>(xc, linW, linB, (float*)d_out, NC_);
}
